// ALiBiMultiheadAttention_87814901334754
// MI455X (gfx1250) — matrix operand order repaired and checked by reading, not compiled, not run on hardware
//
#include <hip/hip_runtime.h>

// ---------------------------------------------------------------------------
// Problem constants (from reference)
// ---------------------------------------------------------------------------
constexpr int E  = 1024;   // embed dim
constexpr int S  = 2048;   // seq len
constexpr int BS = 2;      // batch
constexpr int H  = 16;     // heads
constexpr int D  = 64;     // head dim
constexpr int R  = 8;      // lora rank
constexpr int ROWS = BS * S; // 4096 tokens

typedef __attribute__((ext_vector_type(16))) __bf16 v16bf;
typedef __attribute__((ext_vector_type(8)))  float  v8f;

union Pack32 { uint4 u[2]; v16bf v; };
union Pack16 { uint4 u; __bf16 h[8]; };

// A-fragment / P-fragment: two 8-element (16B) chunks
__device__ inline v16bf frag_from2(const __bf16* c0, const __bf16* c1) {
  Pack32 p;
  p.u[0] = *(const uint4*)c0;
  p.u[1] = *(const uint4*)c1;
  return p.v;
}
// B-fragment: 16 contiguous bf16 (32B)
__device__ inline v16bf frag_from16(const __bf16* c) { return frag_from2(c, c + 8); }

__device__ inline v8f wmma_bf16(v16bf a, v16bf b, v8f c) {
  return __builtin_amdgcn_wmma_f32_16x16x32_bf16(false, a, false, b, (short)0, c,
                                                 false, false);
}

// reductions across the 16 lanes of a half-wave (xor masks stay in-group)
__device__ inline float hmax16(float v) {
  v = fmaxf(v, __shfl_xor(v, 1));
  v = fmaxf(v, __shfl_xor(v, 2));
  v = fmaxf(v, __shfl_xor(v, 4));
  v = fmaxf(v, __shfl_xor(v, 8));
  return v;
}
__device__ inline float hsum16(float v) {
  v += __shfl_xor(v, 1);
  v += __shfl_xor(v, 2);
  v += __shfl_xor(v, 4);
  v += __shfl_xor(v, 8);
  return v;
}
__device__ inline float wsum32(float v) {
  v += __shfl_xor(v, 1);
  v += __shfl_xor(v, 2);
  v += __shfl_xor(v, 4);
  v += __shfl_xor(v, 8);
  v += __shfl_xor(v, 16);
  return v;
}

// ---------------------------------------------------------------------------
// f32 -> bf16 conversion
// ---------------------------------------------------------------------------
__global__ void cvt_f32_bf16(const float* __restrict__ in, __bf16* __restrict__ out,
                             int n) {
  int i = blockIdx.x * blockDim.x + threadIdx.x;
  if (i < n) out[i] = (__bf16)in[i];
}

// ---------------------------------------------------------------------------
// LoRA first stage: xa[row][r] = sum_k x[row,k] * A[r,k]   (f32 input)
// block = 256 threads = 8 waves; wave w computes rank r = w for one row.
// blockIdx.y selects projection (q/k/v).
// ---------------------------------------------------------------------------
__global__ __launch_bounds__(256) void lora_xa_kernel(
    const float* __restrict__ x, const float* A0, const float* A1, const float* A2,
    float* xa0, float* xa1, float* xa2) {
  const int row = blockIdx.x;
  const int sel = blockIdx.y;
  const float* A = (sel == 0) ? A0 : (sel == 1) ? A1 : A2;
  float* xa      = (sel == 0) ? xa0 : (sel == 1) ? xa1 : xa2;
  const int wave = threadIdx.x >> 5, lane = threadIdx.x & 31;
  const float* xr = x + (size_t)row * E;
  const float* ar = A + (size_t)wave * E;
  float s = 0.f;
  for (int k = lane; k < E; k += 32) s += xr[k] * ar[k];
  s = wsum32(s);
  if (lane == 0) xa[(size_t)row * R + wave] = s;
}

// Same but the activation is bf16 (attention output O)
__global__ __launch_bounds__(256) void lora_oa_kernel(
    const __bf16* __restrict__ o, const float* __restrict__ A,
    float* __restrict__ oa) {
  const int row = blockIdx.x;
  const int wave = threadIdx.x >> 5, lane = threadIdx.x & 31;
  const __bf16* xr = o + (size_t)row * E;
  const float* ar = A + (size_t)wave * E;
  float s = 0.f;
  for (int k = lane; k < E; k += 32) s += (float)xr[k] * ar[k];
  s = wsum32(s);
  if (lane == 0) oa[(size_t)row * R + wave] = s;
}

// ---------------------------------------------------------------------------
// 128x128 GEMM mainloop (C = X * W^T over K=1024), bf16 WMMA, f32 accumulate.
// 256 threads = 8 waves arranged 4(M) x 2(N); each wave owns 32x64 of C.
// ---------------------------------------------------------------------------
__device__ inline void gemm128(const __bf16* __restrict__ Xbf,
                               const __bf16* __restrict__ Wbf, int rowBase,
                               int colBase, __bf16 (*As)[32], __bf16 (*Bs)[32],
                               v8f acc[2][4]) {
  const int tid = threadIdx.x;
  const int wave = tid >> 5, lane = tid & 31;
  const int half = lane >> 4, q = lane & 15;
  const int wm = wave >> 1, wn = wave & 1;
  const int lr = tid >> 1, hl = tid & 1;  // cooperative-load row / half

  for (int kk = 0; kk < E; kk += 32) {
    // stage A tile (X rows) and B tile (W rows = output columns)
    {
      const uint4* src = (const uint4*)(Xbf + (size_t)(rowBase + lr) * E + kk + hl * 16);
      uint4* dst = (uint4*)(&As[lr][hl * 16]);
      dst[0] = src[0];
      dst[1] = src[1];
      const uint4* srcb = (const uint4*)(Wbf + (size_t)(colBase + lr) * E + kk + hl * 16);
      uint4* dstb = (uint4*)(&Bs[lr][hl * 16]);
      dstb[0] = srcb[0];
      dstb[1] = srcb[1];
    }
    __syncthreads();

    v16bf a[2], b[4];
#pragma unroll
    for (int i = 0; i < 2; ++i) {
      const int row = wm * 32 + i * 16 + q;
      a[i] = frag_from2(&As[row][half * 16], &As[row][half * 16 + 8]);
    }
#pragma unroll
    for (int j = 0; j < 4; ++j) {
      const int n = wn * 64 + j * 16 + q;
      b[j] = frag_from16(&Bs[n][half * 16]);
    }
#pragma unroll
    for (int i = 0; i < 2; ++i)
#pragma unroll
      for (int j = 0; j < 4; ++j) acc[i][j] = wmma_bf16(a[i], b[j], acc[i][j]);
    __syncthreads();
  }
}

// ---------------------------------------------------------------------------
// QKV projection: out = x @ W^T + b + 0.125 * xa @ Bm^T, emitted as bf16.
// blockIdx.z selects projection.
// ---------------------------------------------------------------------------
__global__ __launch_bounds__(256) void qkv_gemm_kernel(
    const __bf16* __restrict__ Xbf, const __bf16* Wqb, const __bf16* Wkb,
    const __bf16* Wvb, const float* bq, const float* bk, const float* bv,
    const float* xaq, const float* xak, const float* xav, const float* Bq,
    const float* Bk, const float* Bv, __bf16* Qo, __bf16* Ko, __bf16* Vo) {
  __shared__ __align__(16) __bf16 As[128][32];
  __shared__ __align__(16) __bf16 Bs[128][32];

  const __bf16* W;
  const float *bias, *xa, *Bm;
  __bf16* dst;
  switch (blockIdx.z) {
    case 0: W = Wqb; bias = bq; xa = xaq; Bm = Bq; dst = Qo; break;
    case 1: W = Wkb; bias = bk; xa = xak; Bm = Bk; dst = Ko; break;
    default: W = Wvb; bias = bv; xa = xav; Bm = Bv; dst = Vo; break;
  }
  const int rowBase = blockIdx.x * 128, colBase = blockIdx.y * 128;
  v8f acc[2][4] = {};
  gemm128(Xbf, W, rowBase, colBase, As, Bs, acc);

  const int tid = threadIdx.x, wave = tid >> 5, lane = tid & 31;
  const int half = lane >> 4, q = lane & 15;
  const int wm = wave >> 1, wn = wave & 1;
#pragma unroll
  for (int i = 0; i < 2; ++i)
#pragma unroll
    for (int j = 0; j < 4; ++j)
#pragma unroll
      for (int r = 0; r < 8; ++r) {
        const int row = rowBase + wm * 32 + i * 16 + half * 8 + r;
        const int col = colBase + wn * 64 + j * 16 + q;
        float v = acc[i][j][r] + bias[col];
        const float* xr = xa + (size_t)row * R;
        const float* br = Bm + (size_t)col * R;
        float l = 0.f;
#pragma unroll
        for (int rr = 0; rr < R; ++rr) l += xr[rr] * br[rr];
        v += 0.125f * l;
        dst[(size_t)row * E + col] = (__bf16)v;
      }
}

// ---------------------------------------------------------------------------
// Flash attention with ALiBi.  Grid: (S/64, H, B).  128 threads = 4 waves;
// wave w owns query rows [qb*64 + w*16, +16).  j-tiles of 64 staged in LDS.
// ---------------------------------------------------------------------------
__global__ __launch_bounds__(128) void attention_kernel(
    const __bf16* __restrict__ Qbf, const __bf16* __restrict__ Kbf,
    const __bf16* __restrict__ Vbf, __bf16* __restrict__ Obf) {
  __shared__ __align__(16) __bf16 Ks[64][64];      // [j][d]
  __shared__ __align__(16) __bf16 Vt[64][64];      // [d][j] (transposed at load)
  __shared__ __align__(16) __bf16 Ps[4][16][64];   // per-wave P tile

  const int qb = blockIdx.x, h = blockIdx.y, b = blockIdx.z;
  const int tid = threadIdx.x;
  const int w = tid >> 5, lane = tid & 31, half = lane >> 4, q = lane & 15;
  const int jr = tid >> 1, hl = tid & 1;  // cooperative-load row / half
  const float slope = exp2f(-(float)h);
  const int qrowg = qb * 64 + w * 16;  // wave's query row base (within sequence)

  // Q fragments for this wave's 16 rows (d is the WMMA K dimension)
  const __bf16* qp = Qbf + ((size_t)(b * S + qrowg + q)) * E + h * D;
  const v16bf qa0 = frag_from2(qp + half * 16, qp + half * 16 + 8);
  const v16bf qa1 = frag_from2(qp + 32 + half * 16, qp + 32 + half * 16 + 8);

  float m_r[8], l_r[8];
  v8f oacc[4] = {};
#pragma unroll
  for (int r = 0; r < 8; ++r) {
    m_r[r] = -1e30f;
    l_r[r] = 0.f;
  }

  for (int jt = 0; jt < S; jt += 64) {
    // ---- stage K [j][d] and transposed V [d][j] tiles --------------------
    {
      const uint4* ks =
          (const uint4*)(Kbf + ((size_t)(b * S + jt + jr)) * E + h * D + hl * 32);
      uint4* kd = (uint4*)(&Ks[jr][hl * 32]);
#pragma unroll
      for (int c = 0; c < 4; ++c) kd[c] = ks[c];

      const __bf16* vp = Vbf + ((size_t)(b * S + jt + jr)) * E + h * D + hl * 32;
#pragma unroll
      for (int c = 0; c < 4; ++c) {
        Pack16 p;
        p.u = ((const uint4*)vp)[c];
#pragma unroll
        for (int e = 0; e < 8; ++e) Vt[hl * 32 + c * 8 + e][jr] = p.h[e];
      }
      if (jt + 64 < S) {  // prefetch next tile -> global_prefetch_b8
        __builtin_prefetch(
            Kbf + ((size_t)(b * S + jt + 64 + jr)) * E + h * D + hl * 32, 0, 1);
        __builtin_prefetch(
            Vbf + ((size_t)(b * S + jt + 64 + jr)) * E + h * D + hl * 32, 0, 1);
      }
    }
    __syncthreads();

    // ---- scores S = Q K^T * 0.125 + slope*(j - i) ------------------------
    float sc[4][8];
#pragma unroll
    for (int tn = 0; tn < 4; ++tn) {
      v8f c = {};
      const v16bf kb0 = frag_from16(&Ks[tn * 16 + q][half * 16]);
      const v16bf kb1 = frag_from16(&Ks[tn * 16 + q][32 + half * 16]);
      c = wmma_bf16(qa0, kb0, c);
      c = wmma_bf16(qa1, kb1, c);
#pragma unroll
      for (int r = 0; r < 8; ++r)
        sc[tn][r] = c[r] * 0.125f +
                    slope * ((float)(jt + tn * 16 + q) -
                             (float)(qrowg + half * 8 + r));
    }

    // ---- online softmax (rows live in 16-lane halves) --------------------
#pragma unroll
    for (int r = 0; r < 8; ++r) {
      float mx = fmaxf(fmaxf(sc[0][r], sc[1][r]), fmaxf(sc[2][r], sc[3][r]));
      mx = hmax16(mx);
      const float mnew = fmaxf(m_r[r], mx);
      const float alpha = expf(m_r[r] - mnew);
      float sum = 0.f;
#pragma unroll
      for (int tn = 0; tn < 4; ++tn) {
        const float p = expf(sc[tn][r] - mnew);
        sc[tn][r] = p;
        sum += p;
      }
      sum = hsum16(sum);
      l_r[r] = l_r[r] * alpha + sum;
      m_r[r] = mnew;
#pragma unroll
      for (int dt = 0; dt < 4; ++dt) oacc[dt][r] *= alpha;
#pragma unroll
      for (int tn = 0; tn < 4; ++tn)
        Ps[w][half * 8 + r][tn * 16 + q] = (__bf16)sc[tn][r];
    }
    __syncthreads();  // P visible; also: all waves done reading Ks

    // ---- O += P @ V ------------------------------------------------------
    const v16bf pa0 =
        frag_from2(&Ps[w][q][half * 16], &Ps[w][q][half * 16 + 8]);
    const v16bf pa1 =
        frag_from2(&Ps[w][q][32 + half * 16], &Ps[w][q][32 + half * 16 + 8]);
#pragma unroll
    for (int dt = 0; dt < 4; ++dt) {
      const v16bf vb0 = frag_from16(&Vt[dt * 16 + q][half * 16]);
      const v16bf vb1 = frag_from16(&Vt[dt * 16 + q][32 + half * 16]);
      oacc[dt] = wmma_bf16(pa0, vb0, oacc[dt]);
      oacc[dt] = wmma_bf16(pa1, vb1, oacc[dt]);
    }
    __syncthreads();  // before overwriting Ks/Vt/Ps next iteration
  }

  // ---- normalize and store O (bf16, [b*S+q][h*64+d]) ---------------------
#pragma unroll
  for (int dt = 0; dt < 4; ++dt)
#pragma unroll
    for (int r = 0; r < 8; ++r) {
      const int qg = qrowg + half * 8 + r;
      Obf[((size_t)(b * S + qg)) * E + h * D + dt * 16 + q] =
          (__bf16)(oacc[dt][r] / l_r[r]);
    }
}

// ---------------------------------------------------------------------------
// Output projection + residual:
//   out = x + rezero * (O @ Wo^T + bo + 0.125 * oa @ Bo^T)
// ---------------------------------------------------------------------------
__global__ __launch_bounds__(256) void o_gemm_kernel(
    const __bf16* __restrict__ Obf, const __bf16* __restrict__ Wob,
    const float* __restrict__ bo, const float* __restrict__ oa,
    const float* __restrict__ Bo, const float* __restrict__ x,
    const float* __restrict__ rezero, float* __restrict__ out) {
  __shared__ __align__(16) __bf16 As[128][32];
  __shared__ __align__(16) __bf16 Bs[128][32];

  const int rowBase = blockIdx.x * 128, colBase = blockIdx.y * 128;
  v8f acc[2][4] = {};
  gemm128(Obf, Wob, rowBase, colBase, As, Bs, acc);

  const float rz = rezero[0];
  const int tid = threadIdx.x, wave = tid >> 5, lane = tid & 31;
  const int half = lane >> 4, q = lane & 15;
  const int wm = wave >> 1, wn = wave & 1;
#pragma unroll
  for (int i = 0; i < 2; ++i)
#pragma unroll
    for (int j = 0; j < 4; ++j)
#pragma unroll
      for (int r = 0; r < 8; ++r) {
        const int row = rowBase + wm * 32 + i * 16 + half * 8 + r;
        const int col = colBase + wn * 64 + j * 16 + q;
        float v = acc[i][j][r] + bo[col];
        const float* xr = oa + (size_t)row * R;
        const float* br = Bo + (size_t)col * R;
        float l = 0.f;
#pragma unroll
        for (int rr = 0; rr < R; ++rr) l += xr[rr] * br[rr];
        v += 0.125f * l;
        const size_t idx = (size_t)row * E + col;
        out[idx] = x[idx] + rz * v;
      }
}

// ---------------------------------------------------------------------------
// Host launcher
// ---------------------------------------------------------------------------
extern "C" void kernel_launch(void* const* d_in, const int* in_sizes, int n_in,
                              void* d_out, int out_size, void* d_ws, size_t ws_size,
                              hipStream_t stream) {
  const float* x  = (const float*)d_in[0];
  const float* Wq = (const float*)d_in[1];
  const float* bq = (const float*)d_in[2];
  const float* Aq = (const float*)d_in[3];
  const float* Bq = (const float*)d_in[4];
  const float* Wk = (const float*)d_in[5];
  const float* bk = (const float*)d_in[6];
  const float* Ak = (const float*)d_in[7];
  const float* Bk = (const float*)d_in[8];
  const float* Wv = (const float*)d_in[9];
  const float* bv = (const float*)d_in[10];
  const float* Av = (const float*)d_in[11];
  const float* Bv = (const float*)d_in[12];
  const float* Wo = (const float*)d_in[13];
  const float* bo = (const float*)d_in[14];
  const float* Ao = (const float*)d_in[15];
  const float* Bo = (const float*)d_in[16];
  const float* rz = (const float*)d_in[17];
  float* out = (float*)d_out;

  // workspace layout (bytes)
  char* w = (char*)d_ws;
  size_t off = 0;
  auto take = [&](size_t bytes) {
    void* p = w + off;
    off += (bytes + 255) & ~(size_t)255;
    return p;
  };
  __bf16* x_bf  = (__bf16*)take((size_t)ROWS * E * 2);
  __bf16* Wq_bf = (__bf16*)take((size_t)E * E * 2);
  __bf16* Wk_bf = (__bf16*)take((size_t)E * E * 2);
  __bf16* Wv_bf = (__bf16*)take((size_t)E * E * 2);
  __bf16* Wo_bf = (__bf16*)take((size_t)E * E * 2);
  __bf16* Q_bf  = (__bf16*)take((size_t)ROWS * E * 2);
  __bf16* K_bf  = (__bf16*)take((size_t)ROWS * E * 2);
  __bf16* V_bf  = (__bf16*)take((size_t)ROWS * E * 2);
  __bf16* O_bf  = (__bf16*)take((size_t)ROWS * E * 2);
  float* xaq = (float*)take((size_t)ROWS * R * 4);
  float* xak = (float*)take((size_t)ROWS * R * 4);
  float* xav = (float*)take((size_t)ROWS * R * 4);
  float* oa  = (float*)take((size_t)ROWS * R * 4);
  (void)ws_size;
  (void)n_in;
  (void)in_sizes;
  (void)out_size;

  // 1) precision conversion
  const int nx = ROWS * E, nw = E * E;
  cvt_f32_bf16<<<(nx + 255) / 256, 256, 0, stream>>>(x, x_bf, nx);
  cvt_f32_bf16<<<(nw + 255) / 256, 256, 0, stream>>>(Wq, Wq_bf, nw);
  cvt_f32_bf16<<<(nw + 255) / 256, 256, 0, stream>>>(Wk, Wk_bf, nw);
  cvt_f32_bf16<<<(nw + 255) / 256, 256, 0, stream>>>(Wv, Wv_bf, nw);
  cvt_f32_bf16<<<(nw + 255) / 256, 256, 0, stream>>>(Wo, Wo_bf, nw);

  // 2) LoRA first stage for q/k/v
  lora_xa_kernel<<<dim3(ROWS, 3), 256, 0, stream>>>(x, Aq, Ak, Av, xaq, xak, xav);

  // 3) fused QKV projection GEMMs (WMMA bf16)
  qkv_gemm_kernel<<<dim3(ROWS / 128, E / 128, 3), 256, 0, stream>>>(
      x_bf, Wq_bf, Wk_bf, Wv_bf, bq, bk, bv, xaq, xak, xav, Bq, Bk, Bv, Q_bf,
      K_bf, V_bf);

  // 4) flash attention with ALiBi (WMMA bf16, online softmax)
  attention_kernel<<<dim3(S / 64, H, BS), 128, 0, stream>>>(Q_bf, K_bf, V_bf, O_bf);

  // 5) LoRA first stage for output projection
  lora_oa_kernel<<<ROWS, 256, 0, stream>>>(O_bf, Ao, oa);

  // 6) output projection + rezero residual
  o_gemm_kernel<<<dim3(ROWS / 128, E / 128), 256, 0, stream>>>(
      O_bf, Wo_bf, bo, oa, Bo, x, rz, out);
}
